// LSTMDecoder_41635412967990
// MI455X (gfx1250) — compile-verified, matrix operand order checked by reading, not run on hardware
//
#include <hip/hip_runtime.h>

#define E      256
#define B      32
#define S      512
#define TWO_E  512
#define G4     1024          // 4*E gates
#define VOCAB  10000
#define VOCAB_PAD 10016      // next multiple of 16
#define LN_EPS 1e-5f

typedef __attribute__((ext_vector_type(16))) __bf16 v16bf;
typedef __attribute__((ext_vector_type(8)))  __bf16 v8bf;
typedef __attribute__((ext_vector_type(8)))  float  v8f;

__device__ __forceinline__ __bf16 f2bf(float f) {
    unsigned u = __builtin_bit_cast(unsigned, f);
    unsigned r = u + 0x7fffu + ((u >> 16) & 1u);   // round-to-nearest-even
    unsigned short h = (unsigned short)(r >> 16);
    return __builtin_bit_cast(__bf16, h);
}
__device__ __forceinline__ float bf2f(__bf16 x) {
    unsigned short h = __builtin_bit_cast(unsigned short, x);
    return __builtin_bit_cast(float, ((unsigned)h) << 16);
}
__device__ __forceinline__ v16bf cat8(v8bf lo, v8bf hi) {
    return __builtin_shufflevector(lo, hi, 0,1,2,3,4,5,6,7,8,9,10,11,12,13,14,15);
}
__device__ __forceinline__ float sigf(float x) { return 1.0f / (1.0f + __expf(-x)); }

// ---------------------------------------------------------------------------
// Weight packing / loop-invariant precompute (tiny, VALU)
// ---------------------------------------------------------------------------

// GW[k] = sum_j ln_g[j]*w_ih[k,j];  BW[k] = sum_j ln_b[j]*w_ih[k,j] + b_ih[k] + b_hh[k]
__global__ void k_gwbw(const float* __restrict__ w_ih, const float* __restrict__ ln_g,
                       const float* __restrict__ ln_b, const float* __restrict__ b_ih,
                       const float* __restrict__ b_hh, float* __restrict__ GW,
                       float* __restrict__ BW) {
    int k = blockIdx.x * blockDim.x + threadIdx.x;
    if (k >= G4) return;
    float gw = 0.f, bw = 0.f;
    for (int j = 0; j < TWO_E; ++j) {
        float w = w_ih[(size_t)k * TWO_E + j];
        gw += ln_g[j] * w;
        bw += ln_b[j] * w;
    }
    GW[k] = gw;
    BW[k] = bw + b_ih[k] + b_hh[k];
}

// Wxp[n,k] = w_ih[n,k] * ln_g[k]   (bf16, [G4][E])
__global__ void k_pack_wx(const float* __restrict__ w_ih, const float* __restrict__ ln_g,
                          __bf16* __restrict__ Wxp) {
    int i = blockIdx.x * blockDim.x + threadIdx.x;
    if (i >= G4 * E) return;
    int n = i >> 8, k = i & 255;
    Wxp[i] = f2bf(w_ih[(size_t)n * TWO_E + k] * ln_g[k]);
}

// Wcat[n,k] = (k<E) ? w_ih[n,E+k]*ln_g[E+k] : w_hh[n,k-E]   (bf16, [G4][TWO_E])
__global__ void k_pack_wcat(const float* __restrict__ w_ih, const float* __restrict__ w_hh,
                            const float* __restrict__ ln_g, __bf16* __restrict__ Wcat) {
    int i = blockIdx.x * blockDim.x + threadIdx.x;
    if (i >= G4 * TWO_E) return;
    int n = i >> 9, k = i & 511;
    float v = (k < E) ? w_ih[(size_t)n * TWO_E + E + k] * ln_g[E + k]
                      : w_hh[(size_t)n * E + (k - E)];
    Wcat[i] = f2bf(v);
}

// Wouth[n,k] = w_outp[n, E+k]   (bf16, [E][E])
__global__ void k_pack_wouth(const float* __restrict__ w_outp, __bf16* __restrict__ Wouth) {
    int i = blockIdx.x * blockDim.x + threadIdx.x;
    if (i >= E * E) return;
    int n = i >> 8, k = i & 255;
    Wouth[i] = f2bf(w_outp[(size_t)n * TWO_E + E + k]);
}

// Wfcb[n,k] = (n<VOCAB) ? w_fc[n,k] : 0    (bf16, [VOCAB_PAD][E])
__global__ void k_pack_wfc(const float* __restrict__ w_fc, __bf16* __restrict__ Wfcb) {
    int i = blockIdx.x * blockDim.x + threadIdx.x;
    if (i >= VOCAB_PAD * E) return;
    int n = i >> 8, k = i & 255;
    Wfcb[i] = f2bf(n < VOCAB ? w_fc[(size_t)n * E + k] : 0.f);
}

// ctx[b,n] = encoder_out[b,:] . w_enc[n,:]   (softmax over srclen==1 is identity)
__global__ void k_encproj(const float* __restrict__ enc, const float* __restrict__ w_enc,
                          float* __restrict__ CTX) {
    int b = blockIdx.x, n = threadIdx.x;
    float acc = 0.f;
    for (int k = 0; k < E; ++k) acc += enc[b * E + k] * w_enc[(size_t)n * E + k];
    CTX[b * E + n] = acc;
}

// ctxp[b,n] = ctx[b,:] . w_outp[n, 0:E]
__global__ void k_ctxproj(const float* __restrict__ CTX, const float* __restrict__ w_outp,
                          float* __restrict__ CTXP) {
    int b = blockIdx.x, n = threadIdx.x;
    float acc = 0.f;
    for (int k = 0; k < E; ++k) acc += CTX[b * E + k] * w_outp[(size_t)n * TWO_E + k];
    CTXP[b * E + n] = acc;
}

// ---------------------------------------------------------------------------
// Embedding gather (time-major rows r = s*B+b) + per-row LN partial stats
// ---------------------------------------------------------------------------
__global__ __launch_bounds__(256)
void k_embed(const int* __restrict__ tok, const float* __restrict__ embed,
             __bf16* __restrict__ Ax, float* __restrict__ Stats) {
    __shared__ float red0[256];
    __shared__ float red1[256];
    int row = blockIdx.x;                 // row = s*B + b
    int s = row >> 5, b = row & 31;
    int t = tok[b * S + s];
    float v = embed[(size_t)t * E + threadIdx.x];
    Ax[(size_t)row * E + threadIdx.x] = f2bf(v);
    red0[threadIdx.x] = v;
    red1[threadIdx.x] = v * v;
    __syncthreads();
    for (int off = 128; off > 0; off >>= 1) {
        if ((int)threadIdx.x < off) {
            red0[threadIdx.x] += red0[threadIdx.x + off];
            red1[threadIdx.x] += red1[threadIdx.x + off];
        }
        __syncthreads();
    }
    if (threadIdx.x == 0) {
        Stats[row * 2 + 0] = red0[0];
        Stats[row * 2 + 1] = red1[0];
    }
}

// ---------------------------------------------------------------------------
// XGW = Ax [16384,256] @ Wxp^T [256,1024]  -> f32 [16384][1024]   (bf16 WMMA)
// One wave per 16x16 C tile; assumed wave32 fragment layouts per ISA 7.12.2.
// ---------------------------------------------------------------------------
__global__ __launch_bounds__(256)
void k_xgw(const __bf16* __restrict__ Ax, const __bf16* __restrict__ Wxp,
           float* __restrict__ XGW) {
    int lane = threadIdx.x & 31;
    int wv = threadIdx.x >> 5;
    int tile = blockIdx.x * 8 + wv;       // 65536 tiles total
    int tm = tile >> 6;                   // 0..1023
    int tn = tile & 63;                   // 0..63
    int am = tm * 16 + (lane & 15);
    int cn = tn * 16 + (lane & 15);
    int koff = (lane >> 4) * 8;
    const __bf16* arow = Ax  + (size_t)am * E + koff;
    const __bf16* brow = Wxp + (size_t)cn * E + koff;
    v8f acc = {0.f, 0.f, 0.f, 0.f, 0.f, 0.f, 0.f, 0.f};
#pragma unroll
    for (int kt = 0; kt < 8; ++kt) {
        int kb = kt * 32;
        v16bf av = cat8(*(const v8bf*)(arow + kb), *(const v8bf*)(arow + kb + 16));
        v16bf bv = cat8(*(const v8bf*)(brow + kb), *(const v8bf*)(brow + kb + 16));
        acc = __builtin_amdgcn_wmma_f32_16x16x32_bf16(false, av, false, bv,
                                                      (short)0, acc, false, false);
    }
    int mr0 = tm * 16 + (lane >> 4) * 8;
#pragma unroll
    for (int r = 0; r < 8; ++r)
        XGW[(size_t)(mr0 + r) * G4 + cn] = acc[r];
}

// ---------------------------------------------------------------------------
// Persistent recurrence: 1 workgroup, 32 wave32, 512 sequential steps.
// LDS-resident [rs*feed | h] (bf16) and gates (bf16); c in registers;
// weights (Wcat 1MB bf16, Wouth 128KB) streamed from L2.
// ---------------------------------------------------------------------------
__global__ __launch_bounds__(1024, 1)
void k_recur(const float* __restrict__ Stats, const float* __restrict__ XGW,
             const float* __restrict__ GWp,  const float* __restrict__ BWp,
             const float* __restrict__ CTXP, const float* __restrict__ enc,
             const __bf16* __restrict__ Wcat, const __bf16* __restrict__ Wouth,
             __bf16* __restrict__ Ybf) {
    __shared__ __bf16 s_acat[B][TWO_E + 8];   // [b][0:E)=rs*feed, [E:2E)=h
    __shared__ __bf16 s_gates[B][G4 + 8];
    __shared__ float  s_GW[G4];
    __shared__ float  s_BW[G4];
    __shared__ float  s_rs[B];
    __shared__ float  s_nm[B];                // -rs*mu
    __shared__ float  s_sf[B];                // sum(feed)
    __shared__ float  s_sff[B];               // sum(feed^2)

    const int tid  = threadIdx.x;
    const int lane = tid & 31;
    const int wv   = tid >> 5;
    const int bo   = tid >> 5;                // owned batch row (elementwise phases)
    const int e0   = (tid & 31) * 8;

    float creg[8];                            // LSTM cell state, register-resident
#pragma unroll
    for (int j = 0; j < 8; ++j) {
        creg[j] = 0.f;
        s_acat[bo][e0 + j]     = f2bf(0.f);                    // feed0 = 0
        s_acat[bo][E + e0 + j] = f2bf(enc[bo * E + e0 + j]);   // h0 = encoder_out
    }
    if (tid < B) { s_sf[tid] = 0.f; s_sff[tid] = 0.f; }
    s_GW[tid] = GWp[tid];
    s_BW[tid] = BWp[tid];
    __syncthreads();

    for (int st = 0; st < S; ++st) {
        // phase 0: LN stats -> rs, -rs*mu per batch row; reset feed accumulators
        if (tid < B) {
            float sx  = Stats[(st * B + tid) * 2 + 0];
            float sxx = Stats[(st * B + tid) * 2 + 1];
            float mu  = (sx + s_sf[tid]) * (1.f / (float)TWO_E);
            float ms  = (sxx + s_sff[tid]) * (1.f / (float)TWO_E);
            float rs  = rsqrtf(ms - mu * mu + LN_EPS);
            s_rs[tid] = rs;
            s_nm[tid] = -rs * mu;
            s_sf[tid] = 0.f; s_sff[tid] = 0.f;
        }
        __syncthreads();
        {   // prescale feed half by rs[b] (folds LN scale into the GEMM A-matrix)
            float rs = s_rs[bo];
#pragma unroll
            for (int j = 0; j < 8; ++j) {
                int e = e0 + j;
                s_acat[bo][e] = f2bf(bf2f(s_acat[bo][e]) * rs);
            }
        }
        __syncthreads();

        // phase A: gates = [rs*feed | h] @ Wcat^T  + rs*XGW - rs*mu*GW + BW
        // 128 16x16 tiles (M=32,N=1024), 4 per wave, K=512 -> 16 WMMA each
#pragma unroll 1
        for (int i = 0; i < 4; ++i) {
            int t  = wv * 4 + i;
            int tm = t >> 6, tn = t & 63;
            int am = tm * 16 + (lane & 15);
            int cn = tn * 16 + (lane & 15);
            int koff = (lane >> 4) * 8;
            const __bf16* wrow = Wcat + (size_t)cn * TWO_E + koff;
            v8f acc = {0.f, 0.f, 0.f, 0.f, 0.f, 0.f, 0.f, 0.f};
#pragma unroll 1
            for (int kt = 0; kt < 16; ++kt) {
                int kb = kt * 32;
                __builtin_prefetch(wrow + kb + 64, 0, 0);
                v16bf av = cat8(*(const v8bf*)&s_acat[am][kb + koff],
                                *(const v8bf*)&s_acat[am][kb + 16 + koff]);
                v16bf bv = cat8(*(const v8bf*)(wrow + kb),
                                *(const v8bf*)(wrow + kb + 16));
                acc = __builtin_amdgcn_wmma_f32_16x16x32_bf16(false, av, false, bv,
                                                              (short)0, acc, false, false);
            }
            int mr0 = tm * 16 + (lane >> 4) * 8;
#pragma unroll
            for (int r = 0; r < 8; ++r) {
                int b = mr0 + r;
                float g = acc[r]
                        + s_rs[b] * XGW[(size_t)(st * B + b) * G4 + cn]
                        + s_nm[b] * s_GW[cn] + s_BW[cn];
                s_gates[b][cn] = f2bf(g);
            }
        }
        __syncthreads();

        // phase B: LSTM cell (gate order i,f,g,o); h -> s_acat[:,E:2E)
#pragma unroll
        for (int j = 0; j < 8; ++j) {
            int e = e0 + j;
            float ig = sigf (bf2f(s_gates[bo][e]));
            float fg = sigf (bf2f(s_gates[bo][E + e]));
            float gg = tanhf(bf2f(s_gates[bo][2 * E + e]));
            float og = sigf (bf2f(s_gates[bo][3 * E + e]));
            float c  = fg * creg[j] + ig * gg;
            creg[j] = c;
            s_acat[bo][E + e] = f2bf(og * tanhf(c));
        }
        __syncthreads();

        // phase C: out = tanh(ctx_proj + h @ Wouth^T); 32 tiles, 1 per wave
        {
            int tm = wv >> 4, tn = wv & 15;
            int am = tm * 16 + (lane & 15);
            int cn = tn * 16 + (lane & 15);
            int koff = (lane >> 4) * 8;
            const __bf16* wrow = Wouth + (size_t)cn * E + koff;
            v8f acc = {0.f, 0.f, 0.f, 0.f, 0.f, 0.f, 0.f, 0.f};
#pragma unroll
            for (int kt = 0; kt < 8; ++kt) {
                int kb = kt * 32;
                v16bf av = cat8(*(const v8bf*)&s_acat[am][E + kb + koff],
                                *(const v8bf*)&s_acat[am][E + kb + 16 + koff]);
                v16bf bv = cat8(*(const v8bf*)(wrow + kb),
                                *(const v8bf*)(wrow + kb + 16));
                acc = __builtin_amdgcn_wmma_f32_16x16x32_bf16(false, av, false, bv,
                                                              (short)0, acc, false, false);
            }
            int mr0 = tm * 16 + (lane >> 4) * 8;
#pragma unroll
            for (int r = 0; r < 8; ++r) {
                int b = mr0 + r;
                float o = tanhf(acc[r] + CTXP[b * E + cn]);
                __bf16 ob = f2bf(o);
                s_acat[b][cn] = ob;                           // raw feed for next step
                Ybf[(size_t)(st * B + b) * E + cn] = ob;      // decoder output
                atomicAdd(&s_sf[b], o);                       // ds_add_f32
                atomicAdd(&s_sff[b], o * o);
            }
        }
        __syncthreads();
    }
}

// ---------------------------------------------------------------------------
// logits = Y [16384,256] @ Wfcb^T [256,10016] + b_fc  (bf16 WMMA, guarded N)
// Output layout: [B][S][VOCAB] (row m = s*B+b -> out[(b*S+s)*VOCAB + n])
// ---------------------------------------------------------------------------
__global__ __launch_bounds__(256)
void k_fc(const __bf16* __restrict__ Ybf, const __bf16* __restrict__ Wfcb,
          const float* __restrict__ bfc, float* __restrict__ Out) {
    const int NT = VOCAB_PAD / 16;        // 626
    int lane = threadIdx.x & 31;
    int wv = threadIdx.x >> 5;
    int tile = blockIdx.x * 8 + wv;       // exactly 1024*626 tiles
    int tm = tile / NT;
    int tn = tile % NT;
    int am = tm * 16 + (lane & 15);
    int cn = tn * 16 + (lane & 15);
    int koff = (lane >> 4) * 8;
    const __bf16* arow = Ybf  + (size_t)am * E + koff;
    const __bf16* brow = Wfcb + (size_t)cn * E + koff;
    v8f acc = {0.f, 0.f, 0.f, 0.f, 0.f, 0.f, 0.f, 0.f};
#pragma unroll
    for (int kt = 0; kt < 8; ++kt) {
        int kb = kt * 32;
        v16bf av = cat8(*(const v8bf*)(arow + kb), *(const v8bf*)(arow + kb + 16));
        v16bf bv = cat8(*(const v8bf*)(brow + kb), *(const v8bf*)(brow + kb + 16));
        acc = __builtin_amdgcn_wmma_f32_16x16x32_bf16(false, av, false, bv,
                                                      (short)0, acc, false, false);
    }
    if (cn < VOCAB) {
        float bias = bfc[cn];
        int mr0 = tm * 16 + (lane >> 4) * 8;
#pragma unroll
        for (int r = 0; r < 8; ++r) {
            int m = mr0 + r;
            int s = m >> 5, b = m & 31;
            Out[((size_t)b * S + s) * VOCAB + cn] = acc[r] + bias;
        }
    }
}

// ---------------------------------------------------------------------------
extern "C" void kernel_launch(void* const* d_in, const int* in_sizes, int n_in,
                              void* d_out, int out_size, void* d_ws, size_t ws_size,
                              hipStream_t stream) {
    (void)in_sizes; (void)n_in; (void)out_size; (void)ws_size;
    const int*   tok    = (const int*)  d_in[0];
    const float* enc    = (const float*)d_in[1];
    const float* embed  = (const float*)d_in[2];
    const float* w_ih   = (const float*)d_in[3];
    const float* w_hh   = (const float*)d_in[4];
    const float* b_ih   = (const float*)d_in[5];
    const float* b_hh   = (const float*)d_in[6];
    const float* ln_g   = (const float*)d_in[7];
    const float* ln_b   = (const float*)d_in[8];
    /* d_in[9] = w_inp: dead — softmax over srclen==1 is identically 1 */
    const float* w_enc  = (const float*)d_in[10];
    const float* w_outp = (const float*)d_in[11];
    const float* w_fc   = (const float*)d_in[12];
    const float* b_fc   = (const float*)d_in[13];
    float* out = (float*)d_out;

    char* ws = (char*)d_ws;
    size_t off = 0;
    auto carve = [&](size_t bytes) -> char* {
        off = (off + 255) & ~(size_t)255;
        char* p = ws + off;
        off += bytes;
        return p;
    };
    __bf16* Ax    = (__bf16*)carve(sizeof(__bf16) * (size_t)S * B * E);
    float*  Stats = (float*) carve(sizeof(float)  * (size_t)S * B * 2);
    float*  XGW   = (float*) carve(sizeof(float)  * (size_t)S * B * G4);
    __bf16* Wxp   = (__bf16*)carve(sizeof(__bf16) * (size_t)G4 * E);
    __bf16* Wcat  = (__bf16*)carve(sizeof(__bf16) * (size_t)G4 * TWO_E);
    __bf16* Wouth = (__bf16*)carve(sizeof(__bf16) * (size_t)E * E);
    __bf16* Wfcb  = (__bf16*)carve(sizeof(__bf16) * (size_t)VOCAB_PAD * E);
    float*  GW    = (float*) carve(sizeof(float)  * G4);
    float*  BW    = (float*) carve(sizeof(float)  * G4);
    float*  CTX   = (float*) carve(sizeof(float)  * B * E);
    float*  CTXP  = (float*) carve(sizeof(float)  * B * E);
    __bf16* Ybf   = (__bf16*)carve(sizeof(__bf16) * (size_t)S * B * E);

    k_gwbw      <<<4, 256, 0, stream>>>(w_ih, ln_g, ln_b, b_ih, b_hh, GW, BW);
    k_pack_wx   <<<(G4 * E) / 256, 256, 0, stream>>>(w_ih, ln_g, Wxp);
    k_pack_wcat <<<(G4 * TWO_E) / 256, 256, 0, stream>>>(w_ih, w_hh, ln_g, Wcat);
    k_pack_wouth<<<(E * E) / 256, 256, 0, stream>>>(w_outp, Wouth);
    k_pack_wfc  <<<(VOCAB_PAD * E) / 256, 256, 0, stream>>>(w_fc, Wfcb);
    k_encproj   <<<B, 256, 0, stream>>>(enc, w_enc, CTX);
    k_ctxproj   <<<B, 256, 0, stream>>>(CTX, w_outp, CTXP);
    k_embed     <<<S * B, 256, 0, stream>>>(tok, embed, Ax, Stats);
    k_xgw       <<<(S * B / 16) * (G4 / 16) / 8, 256, 0, stream>>>(Ax, Wxp, XGW);
    k_recur     <<<1, 1024, 0, stream>>>(Stats, XGW, GW, BW, CTXP, enc, Wcat, Wouth, Ybf);
    k_fc        <<<(S * B / 16) * (VOCAB_PAD / 16) / 8, 256, 0, stream>>>(Ybf, Wfcb, b_fc, out);
}
